// FeedbackProjection_43224550867485
// MI455X (gfx1250) — matrix operand order repaired and checked by reading, not compiled, not run on hardware
//
#include <hip/hip_runtime.h>
#include <hip/hip_bf16.h>
#include <math.h>

// ---- CDNA5 WMMA / TDM vector types ----
typedef __attribute__((ext_vector_type(16))) __bf16 v16bf;
typedef __attribute__((ext_vector_type(8)))  __bf16 v8bf;
typedef __attribute__((ext_vector_type(8)))  float  v8f;
typedef unsigned int u32x4 __attribute__((ext_vector_type(4)));
typedef int          i32x4 __attribute__((ext_vector_type(4)));
typedef int          i32x8 __attribute__((ext_vector_type(8)));

static constexpr int  BDIM = 4096;
static constexpr int  CDIM = 7;
static constexpr int  DDIM = 1024;
static constexpr long TOK  = (long)BDIM * CDIM;            // 28672 tokens
static constexpr long BCD  = (long)BDIM * CDIM * DDIM;     // 29,360,128 elems

// LDS tile: 128 rows x (64 data + 8 pad) bf16 = 144B rows (conflict-free b128 reads)
static constexpr unsigned TILE_ELEMS = 128 * 72;
static constexpr unsigned TILE_BYTES = TILE_ELEMS * 2;     // 18432

enum { EPI_GELU = 0, EPI_ERR = 1, EPI_BF = 2, EPI_SIG = 3, EPI_UPD = 4 };

__device__ __forceinline__ float gelu_exact(float x) {
  return 0.5f * x * (1.0f + erff(x * 0.70710678118654752440f));
}

__device__ __forceinline__ v16bf cat8(v8bf lo, v8bf hi) {
  v16bf r;
#pragma unroll
  for (int i = 0; i < 8; ++i) { r[i] = lo[i]; r[8 + i] = hi[i]; }
  return r;
}

// ---------------------------------------------------------------------------
// Tensor Data Mover: DMA one 128-row x 128-byte tile (16 units of 8B per row)
// from global into LDS, padding each 32 DWORDs (128B) with 4 DWORDs (16B) to
// produce 144B LDS rows. Descriptor per CDNA5 ISA ch.8 (D# groups 0/1).
//   ldsAddr     : LDS byte address of tile start
//   gptr        : global byte address of tile start
//   strideUnits : tensor row stride in 8-byte units
// ---------------------------------------------------------------------------
__device__ __forceinline__ void tdm_load_tile(unsigned ldsAddr, const void* gptr,
                                              unsigned strideUnits) {
  const unsigned long long ga = (unsigned long long)(uintptr_t)gptr;
  u32x4 g0;
  g0[0] = 1u;                                   // count=1 (valid user descriptor)
  g0[1] = ldsAddr;                              // lds_addr (bytes)
  g0[2] = (unsigned)ga;                         // global_addr[31:0]
  g0[3] = (unsigned)((ga >> 32) & 0x01FFFFFFull) | (2u << 30);  // addr[56:32], type=2
  i32x8 g1;
  // data_size=3 (8B), pad_enable, pad_interval=4 (32 DW = 128B), pad_amount=3 (4 DW = 16B)
  g1[0] = (int)((3u << 16) | (1u << 20) | (4u << 22) | (3u << 25));
  g1[1] = 0;                                    // abar=0, tensor_dim0[15:0]=0
  g1[2] = (int)0x00004000;                      // tensor_dim0=0x40000000 (hi16), tensor_dim1 lo16=0
  g1[3] = (int)(0x4000u | (16u << 16));         // tensor_dim1 hi16, tile_dim0=16 units
  g1[4] = 128;                                  // tile_dim1=128 rows, tile_dim2=0
  g1[5] = (int)strideUnits;                     // tensor_dim0_stride[31:0]
  g1[6] = 0;                                    // stride[47:32]=0, dim1_stride lo=0
  g1[7] = 0;
  i32x4 z4 = {0, 0, 0, 0};
#if defined(__clang_major__) && (__clang_major__ >= 23)
  i32x8 z8 = {0, 0, 0, 0, 0, 0, 0, 0};
  __builtin_amdgcn_tensor_load_to_lds(g0, g1, z4, z4, z8, 0);
#else
  __builtin_amdgcn_tensor_load_to_lds(g0, g1, z4, z4, 0);
#endif
}

// ---------------------------------------------------------------------------
// Generic bf16 WMMA GEMM:  C[M,N] = epi( A[M,K] * Bw[N,K]^T + bias[N] )
//  - 128x128 block tile, 256 threads = 8 waves, each wave 64x32 (4x2 WMMA tiles)
//  - BK = 64, TDM-fed double-buffered LDS, one barrier per K-stage
//  - dual-A (tile-uniform select): for k >= Ksplit read A2 at (k - Ksplit)
//  - blockIdx.z strides (colony GEMMs)
// ---------------------------------------------------------------------------
template <int EPI>
__global__ __launch_bounds__(256)
void gemm_bf16_k(const __bf16* __restrict__ A, const __bf16* __restrict__ A2,
                 long ldA, int Ksplit,
                 const __bf16* __restrict__ Bw,   // [N][K], ld = K
                 const float* __restrict__ bias,  // [N]
                 int M, int N, int K, long ldC,
                 long strideAz, long strideBz, long strideCz, long strideBiasz,
                 __bf16* __restrict__ outBf, float* __restrict__ outF,
                 const float* __restrict__ auxF,    // EPI_ERR: bottom_up
                 const __bf16* __restrict__ auxBf,  // EPI_UPD: gain
                 const float* __restrict__ fsArr,   // EPI_UPD: feedback_strength
                 int firstIter)
{
  (void)M; (void)N;
  __shared__ __bf16 lA[2][TILE_ELEMS];
  __shared__ __bf16 lB[2][TILE_ELEMS];

  const long z = blockIdx.z;
  A    += z * strideAz;
  A2   += z * strideAz;
  Bw   += z * strideBz;
  bias += z * strideBiasz;
  const long czoff = z * strideCz;

  const int t     = threadIdx.x;
  const int lane  = t & 31;
  const int lmod  = lane & 15;
  const int lhalf = lane >> 4;
  const int wave  = t >> 5;
  const int wm    = (wave & 1) * 64;
  const int wn    = (wave >> 1) * 32;

  const long rowBase = (long)blockIdx.y * 128;
  const long colBase = (long)blockIdx.x * 128;

  // scalar (wave-uniform) predicate: only wave 0 drives the TDM
  const bool w0 = ((__builtin_amdgcn_readfirstlane(t) >> 5) == 0);
  const unsigned ldsA0 = (unsigned)(uintptr_t)&lA[0][0];
  const unsigned ldsB0 = (unsigned)(uintptr_t)&lB[0][0];

  const int KT = K / 64;

  auto issue = [&](int kt) {
    const int k0 = kt * 64;
    // tile-uniform dual-A select (Ksplit is a multiple of 64)
    const __bf16* Abase; int kk;
    if (k0 < Ksplit) { Abase = A;  kk = k0; }
    else             { Abase = A2; kk = k0 - Ksplit; }
    const unsigned buf = (unsigned)(kt & 1) * TILE_BYTES;
    tdm_load_tile(ldsA0 + buf, Abase + rowBase * ldA + kk, (unsigned)(ldA >> 2));
    tdm_load_tile(ldsB0 + buf, Bw + colBase * (long)K + k0, (unsigned)(K >> 2));
  };

  if (w0) issue(0);

  v8f acc[4][2] = {};

  for (int kt = 0; kt < KT; ++kt) {
    if (w0) __builtin_amdgcn_s_wait_tensorcnt(0);  // tile kt landed in LDS
    __syncthreads();                               // publish to all waves
    if (w0 && (kt + 1 < KT)) issue(kt + 1);        // DMA next tile during compute

    const __bf16* aT = &lA[kt & 1][0];
    const __bf16* bT = &lB[kt & 1][0];

#pragma unroll
    for (int ks = 0; ks < 64; ks += 32) {
      v16bf af[4], bfm[2];
      // A frag (16x32 bf16): lanes<16 hold K[0..8)+K[16..24), lanes>=16 K[8..16)+K[24..32)
#pragma unroll
      for (int mt = 0; mt < 4; ++mt) {
        const int r = wm + mt * 16 + lmod;
        v8bf lo = *(const v8bf*)(aT + r * 72 + ks + lhalf * 8);
        v8bf hi = *(const v8bf*)(aT + r * 72 + ks + 16 + lhalf * 8);
        af[mt] = cat8(lo, hi);
      }
      // B frag (32x16 bf16): lanes<16 hold K[0..8)+K[16..24), lanes>=16 K[8..16)+K[24..32)
#pragma unroll
      for (int nt = 0; nt < 2; ++nt) {
        const int n = wn + nt * 16 + lmod;
        v8bf lo = *(const v8bf*)(bT + n * 72 + ks + lhalf * 8);
        v8bf hi = *(const v8bf*)(bT + n * 72 + ks + 16 + lhalf * 8);
        bfm[nt] = cat8(lo, hi);
      }
#pragma unroll
      for (int mt = 0; mt < 4; ++mt)
#pragma unroll
        for (int nt = 0; nt < 2; ++nt)
          acc[mt][nt] = __builtin_amdgcn_wmma_f32_16x16x32_bf16(
              false, af[mt], false, bfm[nt], (short)0, acc[mt][nt], false, false);
    }
  }

  // epilogue: C/D layout: lane l -> N = l%16, M = (l/16)*8 + vgpr
  // (per fixed vgpr, half-wave stores are 64B/32B row-contiguous -> coalesced)
#pragma unroll
  for (int mt = 0; mt < 4; ++mt) {
#pragma unroll
    for (int nt = 0; nt < 2; ++nt) {
      const long c  = colBase + wn + nt * 16 + lmod;
      const long r0 = rowBase + wm + mt * 16 + lhalf * 8;
      const float bv = bias[c];
#pragma unroll
      for (int v = 0; v < 8; ++v) {
        const long r   = r0 + v;
        const float x  = acc[mt][nt][v] + bv;
        const long idx = r * ldC + c + czoff;
        if constexpr (EPI == EPI_GELU) {
          outBf[idx] = (__bf16)gelu_exact(x);
        } else if constexpr (EPI == EPI_ERR) {
          const float e = auxF[idx] - x;          // errors = bottom_up - pred
          outBf[idx] = (__bf16)e;
          const float es = e * (1.0f / 3.0f);     // accumulate avg_errors
          outF[idx] = firstIter ? es : (outF[idx] + es);
        } else if constexpr (EPI == EPI_BF) {
          outBf[idx] = (__bf16)x;
        } else if constexpr (EPI == EPI_SIG) {
          outBf[idx] = (__bf16)(1.0f / (1.0f + expf(-x)));
        } else {  // EPI_UPD: current += pe * gain * fs * (5/3); also bf16 copy
          const int colony = (int)(r % 7);
          const float g  = (float)auxBf[idx];
          const float nc = outF[idx] + x * g * fsArr[colony] * (5.0f / 3.0f);
          outF[idx]  = nc;
          outBf[idx] = (__bf16)nc;
        }
      }
    }
  }
}

// Fano mask == ones(7,7) - I  =>  feedback_i = (sum_j fb_j) - fb_i
__global__ __launch_bounds__(256)
void combine_k(const __bf16* __restrict__ fbraw,
               float* __restrict__ fbOutF, __bf16* __restrict__ fbOutBf)
{
  const long i = (long)blockIdx.x * blockDim.x + threadIdx.x;  // over B*D
  const long n = (long)BDIM * DDIM;
  if (i >= n) return;
  const long b = i / DDIM;
  const int  d = (int)(i % DDIM);
  const long base = b * (long)(CDIM * DDIM) + d;
  float v[CDIM]; float s = 0.f;
#pragma unroll
  for (int j = 0; j < CDIM; ++j) { v[j] = (float)fbraw[base + (long)j * DDIM]; s += v[j]; }
#pragma unroll
  for (int j = 0; j < CDIM; ++j) {
    const float f = s - v[j];
    const long idx = base + (long)j * DDIM;
    fbOutF[idx]  = f;
    fbOutBf[idx] = (__bf16)f;
  }
}

__global__ __launch_bounds__(256)
void cvt_k(const float* __restrict__ s, __bf16* __restrict__ d, long n) {
  const long i = (long)blockIdx.x * blockDim.x + threadIdx.x;
  if (i < n) d[i] = (__bf16)s[i];
}

__global__ __launch_bounds__(256)
void init_k(const float* __restrict__ hs, float* __restrict__ cur,
            __bf16* __restrict__ curBf,
            const float* __restrict__ fs, float* __restrict__ fsOut, long n) {
  const long i = (long)blockIdx.x * blockDim.x + threadIdx.x;
  if (i < n) { const float v = hs[i]; cur[i] = v; curBf[i] = (__bf16)v; }
  if (i < CDIM) fsOut[i] = fs[i];
}

// ---------------------------------------------------------------------------
extern "C" void kernel_launch(void* const* d_in, const int* in_sizes, int n_in,
                              void* d_out, int out_size, void* d_ws, size_t ws_size,
                              hipStream_t stream) {
  (void)in_sizes; (void)n_in; (void)out_size; (void)ws_size;
  const float* hs  = (const float*)d_in[0];
  const float* bup = (const float*)d_in[1];
  const float* cbW = (const float*)d_in[2];
  const float* cbB = (const float*)d_in[3];
  const float* p1W = (const float*)d_in[4];
  const float* p1B = (const float*)d_in[5];
  const float* p2W = (const float*)d_in[6];
  const float* p2B = (const float*)d_in[7];
  const float* e1W = (const float*)d_in[8];
  const float* e1B = (const float*)d_in[9];
  const float* e2W = (const float*)d_in[10];
  const float* e2B = (const float*)d_in[11];
  const float* gW  = (const float*)d_in[12];
  const float* gB  = (const float*)d_in[13];
  const float* fs  = (const float*)d_in[14];

  float* outCur = (float*)d_out;        // [B,C,D]
  float* outFb  = outCur + BCD;         // [B,C,D]
  float* outErr = outFb + BCD;          // [B,C,D]
  float* outFs  = outErr + BCD;         // [7]

  // workspace layout (~366 MB)
  char* p = (char*)d_ws;
  __bf16* curBf = (__bf16*)p;  p += BCD * 2;
  __bf16* Hbuf  = (__bf16*)p;  p += BCD * 4;     // tokens x 2048 bf16
  __bf16* gainBf = Hbuf;                          // alias (H dead after GEMM2)
  __bf16* pe1Bf  = Hbuf + BCD;                    // alias, second half
  __bf16* errBf = (__bf16*)p;  p += BCD * 2;
  __bf16* fbRaw = (__bf16*)p;  p += BCD * 2;
  __bf16* fbBf  = (__bf16*)p;  p += BCD * 2;
  __bf16* wp1 = (__bf16*)p; p += 2048l * 1024 * 2;
  __bf16* wp2 = (__bf16*)p; p += 2048l * 1024 * 2;
  __bf16* we1 = (__bf16*)p; p += 1024l * 1024 * 2;
  __bf16* we2 = (__bf16*)p; p += 1024l * 1024 * 2;
  __bf16* wg  = (__bf16*)p; p += 2048l * 1024 * 2;
  __bf16* wcb = (__bf16*)p; p += 7l * 1024 * 1024 * 2;

  const dim3 blk(256);
  auto cg = [](long n) { return dim3((unsigned)((n + 255) / 256)); };

  // one-time f32 -> bf16 conversions (weights + initial state)
  cvt_k<<<cg(2048l * 1024), blk, 0, stream>>>(p1W, wp1, 2048l * 1024);
  cvt_k<<<cg(2048l * 1024), blk, 0, stream>>>(p2W, wp2, 2048l * 1024);
  cvt_k<<<cg(1024l * 1024), blk, 0, stream>>>(e1W, we1, 1024l * 1024);
  cvt_k<<<cg(1024l * 1024), blk, 0, stream>>>(e2W, we2, 1024l * 1024);
  cvt_k<<<cg(2048l * 1024), blk, 0, stream>>>(gW,  wg,  2048l * 1024);
  cvt_k<<<cg(7l * 1024 * 1024), blk, 0, stream>>>(cbW, wcb, 7l * 1024 * 1024);
  init_k<<<cg(BCD), blk, 0, stream>>>(hs, outCur, curBf, fs, outFs, BCD);

  const unsigned MT = (unsigned)(TOK / 128);  // 224 row tiles of tokens

  for (int it = 0; it < 3; ++it) {
    // 1) H = gelu(cur @ p1^T + b1)   [28672 x 2048], K=1024
    gemm_bf16_k<EPI_GELU><<<dim3(16, MT, 1), blk, 0, stream>>>(
        curBf, curBf, 1024, 1024, wp1, p1B, (int)TOK, 2048, 1024, 2048,
        0, 0, 0, 0, Hbuf, nullptr, nullptr, nullptr, nullptr, 0);
    // 2) err = bup - (H @ p2^T + b2); err_sum += err/3
    gemm_bf16_k<EPI_ERR><<<dim3(8, MT, 1), blk, 0, stream>>>(
        Hbuf, Hbuf, 2048, 2048, wp2, p2B, (int)TOK, 1024, 2048, 1024,
        0, 0, 0, 0, errBf, outErr, bup, nullptr, nullptr, it == 0 ? 1 : 0);
    // 3) fbRaw[b,j,:] = cur[b,j,:] @ cbW[j]^T + cb_b[j]   (grid.z = colony)
    gemm_bf16_k<EPI_BF><<<dim3(8, 32, 7), blk, 0, stream>>>(
        curBf, curBf, 7168, 1024, wcb, cbB, BDIM, 1024, 1024, 7168,
        /*Az*/1024, /*Bz*/1024l * 1024, /*Cz*/1024, /*biasz*/1024,
        fbRaw, nullptr, nullptr, nullptr, nullptr, 0);
    // 4) feedback_i = sum_j fb_j - fb_i   (Fano mask == J - I)
    combine_k<<<cg((long)BDIM * DDIM), blk, 0, stream>>>(fbRaw, outFb, fbBf);
    // 5) gain = sigmoid([cur | fb] @ g^T + g_b), dual-A K=2048, split at 1024
    gemm_bf16_k<EPI_SIG><<<dim3(8, MT, 1), blk, 0, stream>>>(
        curBf, fbBf, 1024, 1024, wg, gB, (int)TOK, 1024, 2048, 1024,
        0, 0, 0, 0, gainBf, nullptr, nullptr, nullptr, nullptr, 0);
    // 6) pe1 = gelu(err @ e1^T + e1_b)
    gemm_bf16_k<EPI_GELU><<<dim3(8, MT, 1), blk, 0, stream>>>(
        errBf, errBf, 1024, 1024, we1, e1B, (int)TOK, 1024, 1024, 1024,
        0, 0, 0, 0, pe1Bf, nullptr, nullptr, nullptr, nullptr, 0);
    // 7) cur += (pe1 @ e2^T + e2_b) * gain * fs * (5/3); refresh curBf
    gemm_bf16_k<EPI_UPD><<<dim3(8, MT, 1), blk, 0, stream>>>(
        pe1Bf, pe1Bf, 1024, 1024, we2, e2B, (int)TOK, 1024, 1024, 1024,
        0, 0, 0, 0, curBf, outCur, nullptr, gainBf, fs, 0);
  }
}
